// RGAT_49478023250644
// MI455X (gfx1250) — compile-verified, matrix operand order checked by reading, not run on hardware
//
#include <hip/hip_runtime.h>
#include <hip/hip_bf16.h>

// ---------------------------------------------------------------------------
// RGAT (2 layers) + MLP head for MI455X (gfx1250, wave32, WMMA).
// Dense math: bf16 inputs, v_wmma_f32_16x16x32_bf16, f32 accumulate.
// One wave computes a 16x64 output tile (4 accumulators). The K loop is
// unrolled by 64 with two structurally-alternating fragment register sets
// (copy-free ping-pong), so next-step loads stay in flight while the
// current step's 4 WMMAs execute.
// Sparse math: f32 atomics (segment max / sum / scatter-add), LDS BN reduce.
// ---------------------------------------------------------------------------

typedef __attribute__((ext_vector_type(16))) __bf16 v16bf;
typedef __attribute__((ext_vector_type(8)))  __bf16 v8bf;
typedef __attribute__((ext_vector_type(8)))  float  v8f;

#define NT  5      // edge types T
#define NH  4      // heads
#define DH  64     // head dim
#define CH  256    // channels (IN == HID == H*D)
#define OUTC 153
#define BN_TILES 4 // column tiles (16 cols each) per wave

// ---------------------------- utility kernels ------------------------------

__global__ void k_f32_to_bf16(const float* __restrict__ in, __bf16* __restrict__ out, int n) {
  int i = blockIdx.x * blockDim.x + threadIdx.x;
  if (i < n) out[i] = (__bf16)in[i];
}

__global__ void k_fill_f32(float* __restrict__ p, float v, int n) {
  int i = blockIdx.x * blockDim.x + threadIdx.x;
  if (i < n) p[i] = v;
}

// Bt[n*K + k] = (n < N) ? W[k*N + n] : 0   (f32 -> bf16, transpose, pad N)
__global__ void k_transpose_to_bf16(const float* __restrict__ W, __bf16* __restrict__ Bt,
                                    int K, int N, int Npad) {
  int i = blockIdx.x * blockDim.x + threadIdx.x;
  if (i >= Npad * K) return;
  int n = i / K, k = i - n * K;
  Bt[i] = (n < N) ? (__bf16)W[(size_t)k * N + n] : (__bf16)0.0f;
}

// ------------------------------ WMMA GEMM ----------------------------------
// C[M x N] = A[M x K] (bf16, row-major, M % 16 == 0)
//          @ B (given as Bt = B^T, Npad x K bf16 row-major, Npad >= gridDim.x*64)
//          [+ bias[col]] [+ addend[row*ldc + col]]
// One wave per 16x64 output tile. K % 64 == 0. A and Bt must be readable
// up to 64 bytes past their end (workspace tail pad covers this).
__device__ __forceinline__ v16bf load_frag(const __bf16* row, int k0, int half) {
  // 16-bit A layout (ISA 7.12.2): lanes 0-15 rows M, K 0-7 then 16-23;
  // lanes 16-31 same rows, K 8-15 then 24-31. B fragment mirrored (Bt rows = cols).
  v8bf lo = *(const v8bf*)(row + k0 + half * 8);
  v8bf hi = *(const v8bf*)(row + k0 + 16 + half * 8);
  v16bf v;
#pragma unroll
  for (int i = 0; i < 8; ++i) { v[i] = lo[i]; v[i + 8] = hi[i]; }
  return v;
}

__global__ void __launch_bounds__(32)
k_gemm_bf16(const __bf16* __restrict__ A,
            const __bf16* __restrict__ Bt,
            int K, int N, int ldc,
            float* __restrict__ Cf,
            __bf16* __restrict__ Cbf,
            const float* __restrict__ addend,
            const float* __restrict__ bias) {
  const int lane = threadIdx.x & 31;
  const int c16  = lane & 15;
  const int half = lane >> 4;
  const int m0 = blockIdx.y << 4;
  const int n0 = blockIdx.x << 6;   // 64 columns per wave

  const __bf16* Arow = A + (size_t)(m0 + c16) * K;
  const __bf16* Brow[BN_TILES];
#pragma unroll
  for (int j = 0; j < BN_TILES; ++j)
    Brow[j] = Bt + (size_t)(n0 + j * 16 + c16) * K;

  v8f acc[BN_TILES];
#pragma unroll
  for (int j = 0; j < BN_TILES; ++j)
    acc[j] = (v8f){0.f, 0.f, 0.f, 0.f, 0.f, 0.f, 0.f, 0.f};

  // prologue: fragments for K-step 0 (ping set)
  v16bf a0 = load_frag(Arow, 0, half);
  v16bf b0[BN_TILES];
#pragma unroll
  for (int j = 0; j < BN_TILES; ++j) b0[j] = load_frag(Brow[j], 0, half);

  for (int k0 = 0; k0 < K; k0 += 64) {
    __builtin_prefetch(Arow + k0 + 128, 0, 0);   // speculative, OOB dropped

    // pong set: K-step k0+32 (issued before consuming ping)
    v16bf a1 = load_frag(Arow, k0 + 32, half);
    v16bf b1[BN_TILES];
#pragma unroll
    for (int j = 0; j < BN_TILES; ++j) b1[j] = load_frag(Brow[j], k0 + 32, half);

#pragma unroll
    for (int j = 0; j < BN_TILES; ++j)
      acc[j] = __builtin_amdgcn_wmma_f32_16x16x32_bf16(false, a0, false, b0[j],
                                                       (short)0, acc[j], false, false);

    // ping set reloaded directly (loop-carried, no register rotation copies):
    // K-step k0+64; <=64B OOB on the final iteration (workspace pad covers it)
    a0 = load_frag(Arow, k0 + 64, half);
#pragma unroll
    for (int j = 0; j < BN_TILES; ++j) b0[j] = load_frag(Brow[j], k0 + 64, half);

#pragma unroll
    for (int j = 0; j < BN_TILES; ++j)
      acc[j] = __builtin_amdgcn_wmma_f32_16x16x32_bf16(false, a1, false, b1[j],
                                                       (short)0, acc[j], false, false);
  }

#pragma unroll
  for (int j = 0; j < BN_TILES; ++j) {
    const int colg = n0 + j * 16 + c16;
    if (colg >= N) continue;  // store guard only (EXEC all-1 through the WMMAs)
    const float badd = bias ? bias[colg] : 0.0f;
#pragma unroll
    for (int r = 0; r < 8; ++r) {
      int row = m0 + half * 8 + r;  // C layout: VGPR r -> M = r (+8 for upper half)
      float v = acc[j][r] + badd;
      if (addend) v += addend[(size_t)row * ldc + colg];
      if (Cf)  Cf [(size_t)row * ldc + colg] = v;
      if (Cbf) Cbf[(size_t)row * ldc + colg] = (__bf16)v;
    }
  }
}

// ------------------------- attention logits --------------------------------
// out[t, n, h] = sum_d proj[t, n, h*64 + d] * avec[t, h, d]
__global__ void k_head_dot(const __bf16* __restrict__ proj, const float* __restrict__ avec,
                           float* __restrict__ out, int rows, int strideRows) {
  int i = blockIdx.x * blockDim.x + threadIdx.x;
  int total = NT * rows * NH;
  if (i >= total) return;
  int h = i & (NH - 1);
  int n = (i >> 2) % rows;
  int t = (i >> 2) / rows;
  const __bf16* p = proj + ((size_t)t * strideRows + n) * CH + h * DH;
  const float*  a = avec + (t * NH + h) * DH;
  float s = 0.f;
#pragma unroll 8
  for (int d = 0; d < DH; ++d) s += (float)p[d] * a[d];
  out[((size_t)t * rows + n) * NH + h] = s;
}

// ------------------------------ edge kernels -------------------------------

__device__ inline void atomic_max_f32(float* addr, float v) {
  if (v >= 0.f) atomicMax((int*)addr, __float_as_int(v));
  else          atomicMin((unsigned int*)addr, __float_as_uint(v));
}

__global__ void k_edge_max(const float* __restrict__ el, const float* __restrict__ er,
                           const int* __restrict__ src, const int* __restrict__ dst,
                           const int* __restrict__ et,
                           float* __restrict__ eh, float* __restrict__ m,
                           int E, int Nsrc, int Ndst) {
  int e = blockIdx.x * blockDim.x + threadIdx.x;
  if (e >= E) return;
  int s = src[e], d = dst[e], t = et[e];
  int seg = d * NT + t;
#pragma unroll
  for (int h = 0; h < NH; ++h) {
    float v = el[((size_t)t * Nsrc + s) * NH + h] + er[((size_t)t * Ndst + d) * NH + h];
    v = v > 0.f ? v : 0.2f * v;                       // leaky_relu(0.2)
    eh[(size_t)e * NH + h] = v;
    atomic_max_f32(&m[(size_t)seg * NH + h], v);
  }
}

__global__ void k_edge_expsum(float* __restrict__ eh, const float* __restrict__ m,
                              float* __restrict__ ssum,
                              const int* __restrict__ dst, const int* __restrict__ et, int E) {
  int e = blockIdx.x * blockDim.x + threadIdx.x;
  if (e >= E) return;
  int seg = dst[e] * NT + et[e];
#pragma unroll
  for (int h = 0; h < NH; ++h) {
    float p = __expf(eh[(size_t)e * NH + h] - m[(size_t)seg * NH + h]);
    eh[(size_t)e * NH + h] = p;
    atomicAdd(&ssum[(size_t)seg * NH + h], p);
  }
}

// one 64-thread block per edge; thread j handles columns [4j, 4j+4)
__global__ void __launch_bounds__(64)
k_edge_agg(const float* __restrict__ eh, const float* __restrict__ ssum,
           const __bf16* __restrict__ proj,
           const int* __restrict__ src, const int* __restrict__ dst,
           const int* __restrict__ et,
           float* __restrict__ agg, int E, int Nsrc) {
  int e = blockIdx.x;
  if (e >= E) return;
  int s = src[e], d = dst[e], t = et[e];
  int seg = d * NT + t;
  int c0 = threadIdx.x * 4;
  int h = c0 >> 6;
  float alpha = eh[(size_t)e * NH + h] / ssum[(size_t)seg * NH + h];
  const __bf16* p = proj + ((size_t)t * Nsrc + s) * CH + c0;
  float* ag = agg + (size_t)seg * CH + c0;
#pragma unroll
  for (int j = 0; j < 4; ++j) atomicAdd(&ag[j], alpha * (float)p[j]);
}

// conv[n, c] = sum_t agg[n*T + t, c] + sum_t bc[t, c]
__global__ void k_conv(const float* __restrict__ agg, const float* __restrict__ bc,
                       float* __restrict__ conv, int ND) {
  int i = blockIdx.x * blockDim.x + threadIdx.x;
  if (i >= ND * CH) return;
  int n = i >> 8, c = i & (CH - 1);
  float s = 0.f;
#pragma unroll
  for (int t = 0; t < NT; ++t)
    s += agg[((size_t)n * NT + t) * CH + c] + bc[t * CH + c];
  conv[i] = s;
}

// ------------------------------ batchnorm ----------------------------------

__global__ void __launch_bounds__(256)
k_bn_stats(const float* __restrict__ h, float* __restrict__ meanv,
           float* __restrict__ varv, int rows) {
  int c = blockIdx.x;  // 256 columns
  float s = 0.f, sq = 0.f;
  for (int r = threadIdx.x; r < rows; r += 256) {
    float v = h[(size_t)r * CH + c];
    s += v; sq += v * v;
  }
  __shared__ float sh[256], sh2[256];
  sh[threadIdx.x] = s; sh2[threadIdx.x] = sq;
  __syncthreads();
  for (int off = 128; off > 0; off >>= 1) {
    if (threadIdx.x < off) {
      sh[threadIdx.x]  += sh[threadIdx.x + off];
      sh2[threadIdx.x] += sh2[threadIdx.x + off];
    }
    __syncthreads();
  }
  if (threadIdx.x == 0) {
    float mu = sh[0] / rows;
    meanv[c] = mu;
    varv[c]  = sh2[0] / rows - mu * mu;   // biased var, matches jnp.var
  }
}

// mode 0: ELU, mode 1: ReLU. Writes f32 and/or bf16 copies.
__global__ void k_bn_act(const float* __restrict__ h, const float* __restrict__ meanv,
                         const float* __restrict__ varv, const float* __restrict__ g,
                         const float* __restrict__ b, float* __restrict__ outf,
                         __bf16* __restrict__ outbf, int n, int relu_mode) {
  int i = blockIdx.x * blockDim.x + threadIdx.x;
  if (i >= n) return;
  int c = i & (CH - 1);
  float v = (h[i] - meanv[c]) * rsqrtf(varv[c] + 1e-5f) * g[c] + b[c];
  v = relu_mode ? fmaxf(v, 0.f) : (v > 0.f ? v : (__expf(v) - 1.f));
  if (outf)  outf[i]  = v;
  if (outbf) outbf[i] = (__bf16)v;
}

// ------------------------------ layer driver -------------------------------

static void run_rgat_layer(const __bf16* xin_bf, int Nsrc, int ND, int E,
                           const int* src, const int* dst, const int* et,
                           const __bf16* wct_l, const float* al_l, const float* ar_l,
                           const float* bc_l, const __bf16* wst_l, const float* bskip_l,
                           const float* gamma_l, const float* beta_l,
                           __bf16* projbf, float* elb, float* erb, float* mb, float* sb,
                           float* ehb, float* aggb, float* convb, float* hb,
                           float* meanb, float* varb,
                           float* xout, __bf16* xoutbf, hipStream_t stream) {
  // 1) per-type projection GEMMs (dominant FLOPs) -> bf16 proj
  for (int t = 0; t < NT; ++t)
    k_gemm_bf16<<<dim3(CH / 64, Nsrc / 16), 32, 0, stream>>>(
        xin_bf, wct_l + (size_t)t * CH * CH, CH, CH, CH,
        nullptr, projbf + (size_t)t * Nsrc * CH, nullptr, nullptr);

  // 2) attention logits
  int nel = NT * Nsrc * NH;
  k_head_dot<<<(nel + 255) / 256, 256, 0, stream>>>(projbf, al_l, elb, Nsrc, Nsrc);
  int ner = NT * ND * NH;
  k_head_dot<<<(ner + 255) / 256, 256, 0, stream>>>(projbf, ar_l, erb, ND, Nsrc);

  // 3) init segment buffers
  int nseg = ND * NT * NH;
  k_fill_f32<<<(nseg + 255) / 256, 256, 0, stream>>>(mb, -__builtin_inff(), nseg);
  k_fill_f32<<<(nseg + 255) / 256, 256, 0, stream>>>(sb, 0.f, nseg);
  int nagg = ND * NT * CH;
  k_fill_f32<<<(nagg + 255) / 256, 256, 0, stream>>>(aggb, 0.f, nagg);

  // 4) edge softmax + aggregation
  k_edge_max<<<(E + 255) / 256, 256, 0, stream>>>(elb, erb, src, dst, et, ehb, mb, E, Nsrc, ND);
  k_edge_expsum<<<(E + 255) / 256, 256, 0, stream>>>(ehb, mb, sb, dst, et, E);
  k_edge_agg<<<E, 64, 0, stream>>>(ehb, sb, projbf, src, dst, et, aggb, E, Nsrc);
  k_conv<<<(ND * CH + 255) / 256, 256, 0, stream>>>(aggb, bc_l, convb, ND);

  // 5) skip GEMM fused with conv + bias
  k_gemm_bf16<<<dim3(CH / 64, ND / 16), 32, 0, stream>>>(
      xin_bf, wst_l, CH, CH, CH, hb, nullptr, convb, bskip_l);

  // 6) batchnorm + ELU, emit f32 + bf16 for next layer
  k_bn_stats<<<CH, 256, 0, stream>>>(hb, meanb, varb, ND);
  k_bn_act<<<(ND * CH + 255) / 256, 256, 0, stream>>>(
      hb, meanb, varb, gamma_l, beta_l, xout, xoutbf, ND * CH, 0);
}

// ------------------------------- entry point -------------------------------

extern "C" void kernel_launch(void* const* d_in, const int* in_sizes, int n_in,
                              void* d_out, int out_size, void* d_ws, size_t ws_size,
                              hipStream_t stream) {
  (void)n_in; (void)out_size; (void)ws_size;

  const int N0  = in_sizes[0] / CH;   // 100000
  const int E0  = in_sizes[1];        // 400000
  const int E1  = in_sizes[4];        // 80000
  const int ND0 = 20000;              // n_dst0 (fixed python scalar)
  const int ND1 = 4000;               // n_dst1 (fixed python scalar)

  const float* x     = (const float*)d_in[0];
  const int*   src0  = (const int*)d_in[1];
  const int*   dst0  = (const int*)d_in[2];
  const int*   et0   = (const int*)d_in[3];
  const int*   src1  = (const int*)d_in[4];
  const int*   dst1  = (const int*)d_in[5];
  const int*   et1   = (const int*)d_in[6];
  const float* Wc    = (const float*)d_in[9];
  const float* al    = (const float*)d_in[10];
  const float* ar    = (const float*)d_in[11];
  const float* bc    = (const float*)d_in[12];
  const float* Wskip = (const float*)d_in[13];
  const float* bskip = (const float*)d_in[14];
  const float* gamma = (const float*)d_in[15];
  const float* beta  = (const float*)d_in[16];
  const float* W1    = (const float*)d_in[17];
  const float* b1    = (const float*)d_in[18];
  const float* gm    = (const float*)d_in[19];
  const float* bm    = (const float*)d_in[20];
  const float* W2    = (const float*)d_in[21];
  const float* b2    = (const float*)d_in[22];
  float* out = (float*)d_out;

  // --- workspace bump allocator (256B aligned) ---
  char* ws = (char*)d_ws;
  size_t off = 0;
  auto carve = [&](size_t bytes) -> void* {
    void* p = ws + off;
    off = (off + bytes + 255) & ~(size_t)255;
    return p;
  };

  __bf16* wct    = (__bf16*)carve((size_t)2 * NT * CH * CH * 2);
  __bf16* wst    = (__bf16*)carve((size_t)2 * CH * CH * 2);
  __bf16* w1t    = (__bf16*)carve((size_t)CH * CH * 2);
  __bf16* w2t    = (__bf16*)carve((size_t)192 * CH * 2);   // padded to 3x64 col tiles
  __bf16* xbf    = (__bf16*)carve((size_t)N0 * CH * 2);
  __bf16* projbf = (__bf16*)carve((size_t)NT * N0 * CH * 2);
  float*  elb    = (float*)carve((size_t)NT * N0 * NH * 4);
  float*  erb    = (float*)carve((size_t)NT * ND0 * NH * 4);
  float*  mb     = (float*)carve((size_t)ND0 * NT * NH * 4);
  float*  sb     = (float*)carve((size_t)ND0 * NT * NH * 4);
  float*  ehb    = (float*)carve((size_t)E0 * NH * 4);
  float*  aggb   = (float*)carve((size_t)ND0 * NT * CH * 4);
  float*  convb  = (float*)carve((size_t)ND0 * CH * 4);
  float*  hb     = (float*)carve((size_t)ND0 * CH * 4);
  float*  meanb  = (float*)carve(CH * 4);
  float*  varb   = (float*)carve(CH * 4);
  float*  xA     = (float*)carve((size_t)ND0 * CH * 4);
  __bf16* xAbf   = (__bf16*)carve((size_t)ND0 * CH * 2);
  float*  xB     = (float*)carve((size_t)ND1 * CH * 4);
  __bf16* xBbf   = (__bf16*)carve((size_t)ND1 * CH * 2);
  __bf16* h1bf   = (__bf16*)carve((size_t)ND1 * CH * 2);
  (void)carve(256);                                        // tail pad for pipelined OOB reads

  // --- weight prep: transpose + bf16 convert (tiny) ---
  const int wgrid = (CH * CH + 255) / 256;
  for (int L = 0; L < 2; ++L)
    for (int t = 0; t < NT; ++t)
      k_transpose_to_bf16<<<wgrid, 256, 0, stream>>>(
          Wc + ((size_t)L * NT + t) * CH * CH,
          wct + ((size_t)L * NT + t) * CH * CH, CH, CH, CH);
  for (int L = 0; L < 2; ++L)
    k_transpose_to_bf16<<<wgrid, 256, 0, stream>>>(
        Wskip + (size_t)L * CH * CH, wst + (size_t)L * CH * CH, CH, CH, CH);
  k_transpose_to_bf16<<<wgrid, 256, 0, stream>>>(W1, w1t, CH, CH, CH);
  k_transpose_to_bf16<<<(192 * CH + 255) / 256, 256, 0, stream>>>(W2, w2t, CH, OUTC, 192);
  k_f32_to_bf16<<<(N0 * CH + 255) / 256, 256, 0, stream>>>(x, xbf, N0 * CH);

  // --- layer 0 ---
  run_rgat_layer(xbf, N0, ND0, E0, src0, dst0, et0,
                 wct, al, ar, bc, wst, bskip, gamma, beta,
                 projbf, elb, erb, mb, sb, ehb, aggb, convb, hb, meanb, varb,
                 xA, xAbf, stream);

  // --- layer 1 (reuses all scratch; strictly smaller sizes) ---
  run_rgat_layer(xAbf, ND0, ND1, E1, src1, dst1, et1,
                 wct + (size_t)NT * CH * CH,
                 al + NT * NH * DH, ar + NT * NH * DH,
                 bc + NT * CH,
                 wst + (size_t)CH * CH, bskip + CH, gamma + CH, beta + CH,
                 projbf, elb, erb, mb, sb, ehb, aggb, convb, hb, meanb, varb,
                 xB, xBbf, stream);

  // --- MLP head ---
  k_gemm_bf16<<<dim3(CH / 64, ND1 / 16), 32, 0, stream>>>(
      xBbf, w1t, CH, CH, CH, hb, nullptr, nullptr, b1);
  k_bn_stats<<<CH, 256, 0, stream>>>(hb, meanb, varb, ND1);
  k_bn_act<<<(ND1 * CH + 255) / 256, 256, 0, stream>>>(
      hb, meanb, varb, gm, bm, nullptr, h1bf, ND1 * CH, 1);
  k_gemm_bf16<<<dim3(3, ND1 / 16), 32, 0, stream>>>(
      h1bf, w2t, CH, OUTC, OUTC, out, nullptr, nullptr, b2);
}
